// FouriDecoder_283467842833
// MI455X (gfx1250) — matrix operand order repaired and checked by reading, not compiled, not run on hardware
//
#include <hip/hip_runtime.h>

#define DD 512
#define SS 1024
#define BB 4
#define HH 8
#define LL 6

typedef __attribute__((ext_vector_type(16))) __bf16 v16bf;
typedef __attribute__((ext_vector_type(8)))  float  v8f;
typedef unsigned int v4u  __attribute__((ext_vector_type(4)));
typedef int          v8i_t __attribute__((ext_vector_type(8)));
typedef int          v4i_t __attribute__((ext_vector_type(4)));

typedef unsigned char __attribute__((address_space(3))) lds_char;

__device__ __forceinline__ unsigned short f2bf(float f) {
    unsigned int u = __float_as_uint(f);
    unsigned int r = (u + 0x7FFFu + ((u >> 16) & 1u)) >> 16;
    return (unsigned short)r;
}

__device__ __forceinline__ unsigned int lds_off(const void* p) {
    return (unsigned int)(unsigned long long)(lds_char*)p;
}

// Issue one TDM 2D-tile load: 128 rows x 32 bf16 (64B) per row, row stride
// `strideElems`, global tile start `gptr`, LDS dest byte-offset `ldsOff`.
// D# per CDNA5 ISA ch.8: group0 {count=1, lds_addr, global_addr, type=2},
// group1 {data_size=2B, tensor dims (large), tile_dim0=32, tile_dim1=128,
// tensor_dim0_stride}.  Remaining groups zero (2D tensor).
__device__ __forceinline__ void tdm_load_tile_128x32(
    unsigned int ldsOff, const unsigned short* gptr, unsigned int strideElems)
{
    unsigned long long ga = (unsigned long long)gptr;
    v4u g0;
    g0.x = 1u;                                            // count=1 (valid user D#)
    g0.y = ldsOff;                                        // lds_addr
    g0.z = (unsigned int)ga;                              // global_addr[31:0]
    g0.w = (unsigned int)((ga >> 32) & 0x01FFFFFFull)     // global_addr[56:32]
           | 0x80000000u;                                 // type=2 ("image")
    v8i_t g1;
    g1[0] = 0x00010000;        // workgroup_mask=0, data_size=1 (2 bytes)
    g1[1] = 0;                 // no atomic barrier; tensor_dim0[15:0]=0
    g1[2] = 0x00000010;        // tensor_dim0=0x100000 (huge, no OOB); td1 lo=0
    g1[3] = (32 << 16) | 0x10; // tile_dim0=32 elems; tensor_dim1=0x100000
    g1[4] = 128;               // tile_dim1=128 rows, tile_dim2=0
    g1[5] = (int)strideElems;  // tensor_dim0_stride[31:0] (data_size units)
    g1[6] = 0;                 // stride hi / dim1_stride lo
    g1[7] = 0;
    v4i_t gz4 = {0, 0, 0, 0};
    v8i_t gz8 = {0, 0, 0, 0, 0, 0, 0, 0};
    __builtin_amdgcn_tensor_load_to_lds(g0, g1, gz4, gz4, gz8, 0);
}

// ---------------------------------------------------------------------------
// bf16 WMMA GEMM with TDM-staged LDS tiles.
//   A: row-major [M][K]   (lda row stride),   batched by aStride
//   B: col-major, stored as B^T row-major [N][K] (ldb = per-col stride)
//   C: transC=0 -> C[row*ldc+col] ; transC=1 -> C[col*ldc+row]
// 256 threads = 8 waves; WG tile 128x128; wave tile 64x32; K-step 32.
// Optional second pass (A2,B2) accumulates into same accumulators.
// ---------------------------------------------------------------------------
__global__ __launch_bounds__(256) void gemm_bf16_wmma(
    const unsigned short* __restrict__ A,  long lda, long aStride,
    const unsigned short* __restrict__ Bm, long ldb, long bStride,
    const unsigned short* __restrict__ A2, const unsigned short* __restrict__ B2,
    float* __restrict__ Cf, unsigned short* __restrict__ Cb, long ldc, int transC,
    int zDiv, long cOuter, long cInner,
    const float* __restrict__ bias, int act, int K)
{
    __shared__ __align__(16) unsigned short ldsbuf[2 * 128 * 32]; // A tile | B tile
#define LA(m, k) ldsbuf[(m) * 32 + (k)]
#define LB(n, k) ldsbuf[4096 + (n) * 32 + (k)]

    const int t      = threadIdx.x;
    const int lane   = t & 31;
    const int waveId = t >> 5;
    const int wm     = (waveId & 1) * 64;
    const int wn     = (waveId >> 1) * 32;
    const long rowBase = (long)blockIdx.y * 128;
    const long colBase = (long)blockIdx.x * 128;
    const int z = blockIdx.z;

    const unsigned int ldsA = lds_off(&ldsbuf[0]);
    const unsigned int ldsB = ldsA + 8192;

    const v8f vzero = {0.f, 0.f, 0.f, 0.f, 0.f, 0.f, 0.f, 0.f};
    v8f acc[4][2];
#pragma unroll
    for (int mt = 0; mt < 4; ++mt)
#pragma unroll
        for (int nt = 0; nt < 2; ++nt) acc[mt][nt] = vzero;

    const int npass = (A2 != nullptr) ? 2 : 1;
    for (int pass = 0; pass < npass; ++pass) {
        const unsigned short* Ab = (pass ? A2 : A)  + (long)z * aStride;
        const unsigned short* Bb = (pass ? B2 : Bm) + (long)z * bStride;
        for (int kb = 0; kb < K; kb += 32) {
            __syncthreads();                       // prior LDS reads done
            if (waveId == 0) {                     // one wave drives the TDM
                tdm_load_tile_128x32(ldsA, Ab + rowBase * lda + kb, (unsigned int)lda);
                tdm_load_tile_128x32(ldsB, Bb + colBase * ldb + kb, (unsigned int)ldb);
                __builtin_amdgcn_s_wait_tensorcnt(0);
            }
            __syncthreads();                       // tiles visible to all waves

            union FragU { v16bf v; unsigned int u[8]; };
            const int hlf = lane >> 4;
            const int l16 = lane & 15;

            FragU fb0, fb1;
#pragma unroll
            for (int vv = 0; vv < 8; ++vv) {
                int kk = (vv < 4 ? 2 * vv : 16 + 2 * (vv - 4)) + 8 * hlf;  // B 32x16: v0-3 -> K 0-7 (+8*half), v4-7 -> K 16-23 (+8*half)
                fb0.u[vv] = *(const unsigned int*)&LB(wn + l16, kk);
                fb1.u[vv] = *(const unsigned int*)&LB(wn + 16 + l16, kk);
            }
#pragma unroll
            for (int mt = 0; mt < 4; ++mt) {
                FragU fa;
                int mrow = wm + mt * 16 + l16;
#pragma unroll
                for (int vv = 0; vv < 8; ++vv) {
                    // A 16x32: v0-3 -> K 0-7 (+8*half), v4-7 -> K 16-23 (+8*half)
                    int kk = (vv < 4 ? 2 * vv : 16 + 2 * (vv - 4)) + 8 * hlf;
                    fa.u[vv] = *(const unsigned int*)&LA(mrow, kk);
                }
                acc[mt][0] = __builtin_amdgcn_wmma_f32_16x16x32_bf16(
                    false, fa.v, false, fb0.v, (short)0, acc[mt][0], false, false);
                acc[mt][1] = __builtin_amdgcn_wmma_f32_16x16x32_bf16(
                    false, fa.v, false, fb1.v, (short)0, acc[mt][1], false, false);
            }
        }
    }

    // ---- epilogue ----
    const long cOff = (long)(z / zDiv) * cOuter + (long)(z % zDiv) * cInner;
    const int hlf = lane >> 4;
    const int l16 = lane & 15;
    for (int mt = 0; mt < 4; ++mt) {
        for (int nt = 0; nt < 2; ++nt) {
#pragma unroll
            for (int r = 0; r < 8; ++r) {
                long gr   = rowBase + wm + mt * 16 + r + 8 * hlf;
                long gcol = colBase + wn + nt * 16 + l16;
                float val = acc[mt][nt][r];
                if (bias) val += bias[gcol];
                if (act == 1) {  // SELU
                    const float kS = 1.0507009873554805f, kA = 1.6732632423543772f;
                    val = (val > 0.f) ? kS * val : kS * kA * (__expf(val) - 1.f);
                }
                long o = transC ? (cOff + gcol * ldc + gr)
                                : (cOff + gr * ldc + gcol);
                if (Cf) Cf[o] = val;
                if (Cb) Cb[o] = f2bf(val);
            }
        }
    }
#undef LA
#undef LB
}

// ---------------------------------------------------------------------------
// Setup / elementwise kernels
// ---------------------------------------------------------------------------
__global__ void init_kernel(const float* __restrict__ xd, const float* __restrict__ xe,
                            float* __restrict__ x, unsigned short* __restrict__ xbf,
                            unsigned short* __restrict__ xebf) {
    long i = (long)blockIdx.x * 256 + threadIdx.x;
    if (i >= (long)BB * SS * DD) return;
    float v = xd[i];
    x[i] = v;
    xbf[i] = f2bf(v);
    xebf[i] = f2bf(xe[i]);
}

// Ddft^T: [1024 rows j][512 cols d]; rows 0..511 = C_D (cos), 512..1023 = S_D (sin)
__global__ void fill_ddft_t(unsigned short* __restrict__ D) {
    long i = (long)blockIdx.x * 256 + threadIdx.x;
    if (i >= (long)1024 * 512) return;
    int r = (int)(i >> 9), d = (int)(i & 511);
    float val;
    if (r < 512) {
        int m = (r * d) & 511;
        val = cosf(6.283185307179586f * (float)m / 512.f);
    } else {
        int m = ((r - 512) * d) & 511;
        val = sinf(6.283185307179586f * (float)m / 512.f);
    }
    D[i] = f2bf(val);
}

__global__ void fill_cs(unsigned short* __restrict__ Cs, unsigned short* __restrict__ NSs) {
    long i = (long)blockIdx.x * 256 + threadIdx.x;
    if (i >= (long)1024 * 1024) return;
    int r = (int)(i >> 10), c = (int)(i & 1023);
    int m = (r * c) & 1023;
    float ang = 6.283185307179586f * (float)m / 1024.f;
    Cs[i]  = f2bf(cosf(ang));
    NSs[i] = f2bf(-sinf(ang));
}

// LayerNorm(x + add) -> x (f32) and bf16 copy; one 512-wide row per block
__global__ __launch_bounds__(256) void ln_add_kernel(
    const float* __restrict__ xin, const float* __restrict__ addv,
    const float* __restrict__ w, const float* __restrict__ b,
    float* __restrict__ xout, unsigned short* __restrict__ xbf)
{
    __shared__ float red[256];
    const int row = blockIdx.x, t = threadIdx.x;
    const long base = (long)row * DD;
    float v0 = xin[base + t]       + addv[base + t];
    float v1 = xin[base + 256 + t] + addv[base + 256 + t];
    red[t] = v0 + v1;
    __syncthreads();
    for (int s2 = 128; s2 > 0; s2 >>= 1) { if (t < s2) red[t] += red[t + s2]; __syncthreads(); }
    float mean = red[0] * (1.f / DD);
    __syncthreads();
    float d0 = v0 - mean, d1 = v1 - mean;
    red[t] = d0 * d0 + d1 * d1;
    __syncthreads();
    for (int s2 = 128; s2 > 0; s2 >>= 1) { if (t < s2) red[t] += red[t + s2]; __syncthreads(); }
    float rstd = rsqrtf(red[0] * (1.f / DD) + 1e-5f);
    float o0 = d0 * rstd * w[t]       + b[t];
    float o1 = d1 * rstd * w[256 + t] + b[256 + t];
    xout[base + t] = o0;       xout[base + 256 + t] = o1;
    xbf[base + t]  = f2bf(o0); xbf[base + 256 + t]  = f2bf(o1);
}

// softmax over head dim of G[b*S+s][h*D+e] (optional), write bf16:
//   transOut=0 -> [b,h,s,e]   transOut=1 -> [b,h,e,s]
__global__ __launch_bounds__(256) void head_softmax(
    const float* __restrict__ G, unsigned short* __restrict__ out,
    int doSoftmax, int transOut)
{
    long idx = (long)blockIdx.x * 256 + threadIdx.x;
    if (idx >= (long)BB * SS * DD) return;
    int e = (int)(idx & 511);
    long bs = idx >> 9;
    int b = (int)(bs >> 10), s = (int)(bs & 1023);
    float v[HH];
#pragma unroll
    for (int h = 0; h < HH; ++h) v[h] = G[bs * (HH * DD) + (long)h * DD + e];
    if (doSoftmax) {
        float mx = v[0];
#pragma unroll
        for (int h = 1; h < HH; ++h) mx = fmaxf(mx, v[h]);
        float sum = 0.f;
#pragma unroll
        for (int h = 0; h < HH; ++h) { v[h] = __expf(v[h] - mx); sum += v[h]; }
        float inv = 1.f / sum;
#pragma unroll
        for (int h = 0; h < HH; ++h) v[h] *= inv;
    }
#pragma unroll
    for (int h = 0; h < HH; ++h) {
        long o = transOut ? (((long)(b * HH + h) * DD + e) * SS + s)
                          : (((long)(b * HH + h) * SS + s) * DD + e);
        out[o] = f2bf(v[h]);
    }
}

// W [H,D,E] f32 -> Wp^T [H*E][D] bf16 : Wp[(h*E+e)][d] = W[h][d][e]
__global__ void pack_headw_t(const float* __restrict__ W, unsigned short* __restrict__ Wp) {
    long idx = (long)blockIdx.x * 256 + threadIdx.x;
    if (idx >= (long)HH * DD * DD) return;
    long rp = idx >> 9, d = idx & 511;
    long h = rp >> 9, e = rp & 511;
    Wp[idx] = f2bf(W[(h * DD + d) * DD + e]);
}

// Wo [D*H, E] (row=d*H+h) f32 -> Wop^T [E][H*D] bf16 : Wop[e][h*D+d] = Wo[d*H+h][e]
__global__ void pack_wo_t(const float* __restrict__ W, unsigned short* __restrict__ Wp) {
    long idx = (long)blockIdx.x * 256 + threadIdx.x;
    if (idx >= (long)HH * DD * DD) return;
    long e = idx >> 12, c = idx & 4095;
    long h = c >> 9, d = c & 511;
    Wp[idx] = f2bf(W[(d * HH + h) * DD + e]);
}

// 512x512 f32 -> transposed bf16: out[n][k] = in[k][n]
__global__ void pack_t512(const float* __restrict__ src, unsigned short* __restrict__ dst) {
    long idx = (long)blockIdx.x * 256 + threadIdx.x;
    if (idx >= (long)DD * DD) return;
    long n = idx >> 9, k = idx & 511;
    dst[idx] = f2bf(src[k * DD + n]);
}

__global__ void copy_out_kernel(const float* __restrict__ x, float* __restrict__ out, long n) {
    long i = (long)blockIdx.x * 256 + threadIdx.x;
    if (i < n) out[i] = x[i];
}

// ---------------------------------------------------------------------------
extern "C" void kernel_launch(void* const* d_in, const int* in_sizes, int n_in,
                              void* d_out, int out_size, void* d_ws, size_t ws_size,
                              hipStream_t stream)
{
    (void)in_sizes; (void)n_in; (void)out_size; (void)ws_size;
    const float* x_enc = (const float*)d_in[0];
    const float* x_dec = (const float*)d_in[1];
    const float* Wq = (const float*)d_in[2];
    const float* bq = (const float*)d_in[3];
    const float* Wk = (const float*)d_in[4];
    const float* bk = (const float*)d_in[5];
    const float* Wv = (const float*)d_in[6];
    const float* bv = (const float*)d_in[7];
    const float* Wo = (const float*)d_in[8];
    const float* bo = (const float*)d_in[9];
    const float* ln1w = (const float*)d_in[10];
    const float* ln1b = (const float*)d_in[11];
    const float* ln2w = (const float*)d_in[12];
    const float* ln2b = (const float*)d_in[13];
    const float* W1 = (const float*)d_in[14];
    const float* b1 = (const float*)d_in[15];
    const float* W2 = (const float*)d_in[16];
    const float* b2 = (const float*)d_in[17];
    const float* ln3w = (const float*)d_in[18];
    const float* ln3b = (const float*)d_in[19];

    char* wsp = (char*)d_ws;
    size_t off = 0;
    auto alloc = [&](size_t bytes) -> void* {
        void* p = wsp + off;
        off = (off + bytes + 255) & ~(size_t)255;
        return p;
    };
    const long NTOK = (long)BB * SS;  // 4096

    float*          x     = (float*)alloc(NTOK * DD * 4);
    unsigned short* xbf   = (unsigned short*)alloc(NTOK * DD * 2);
    unsigned short* xebf  = (unsigned short*)alloc(NTOK * DD * 2);
    unsigned short* DdftT = (unsigned short*)alloc((long)1024 * 512 * 2);
    unsigned short* Cs    = (unsigned short*)alloc((long)1024 * 1024 * 2);
    unsigned short* NSs   = (unsigned short*)alloc((long)1024 * 1024 * 2);
    unsigned short* Tt    = (unsigned short*)alloc((long)1024 * NTOK * 2);  // [q][token]
    float*          addb  = (float*)alloc(NTOK * DD * 4);
    float*          G     = (float*)alloc(NTOK * 4096 * 4);                 // 64MB scratch
    unsigned short* attbf = (unsigned short*)G;                              // alias: 32MB
    unsigned short* gcbfT = (unsigned short*)((char*)G + (size_t)32 * 1024 * 1024); // [b,h,e,d]
    unsigned short* Qbf   = (unsigned short*)alloc((long)BB * HH * SS * DD * 2);    // [b,h,s,e]
    unsigned short* KbfT  = (unsigned short*)alloc((long)BB * HH * SS * DD * 2);    // [b,h,e,s]
    unsigned short* VbfT  = (unsigned short*)alloc((long)BB * HH * SS * DD * 2);    // [b,h,e,s]
    unsigned short* WqpT  = (unsigned short*)alloc((long)4096 * DD * 2);
    unsigned short* WkpT  = (unsigned short*)alloc((long)4096 * DD * 2);
    unsigned short* WvpT  = (unsigned short*)alloc((long)4096 * DD * 2);
    unsigned short* WopT  = (unsigned short*)alloc((long)DD * 4096 * 2);
    unsigned short* W1pT  = (unsigned short*)alloc((long)DD * DD * 2);
    unsigned short* W2pT  = (unsigned short*)alloc((long)DD * DD * 2);
    unsigned short* h1bf  = (unsigned short*)alloc(NTOK * DD * 2);

    auto gemm = [&](const unsigned short* A, long lda, long aS,
                    const unsigned short* Bmat, long ldb, long bS,
                    const unsigned short* A2, const unsigned short* B2,
                    float* Cf, unsigned short* Cb, long ldc, int transC,
                    int zDiv, long cOuter, long cInner,
                    const float* bias, int act, int M, int N, int K, int Z) {
        dim3 grid(N / 128, M / 128, Z);
        gemm_bf16_wmma<<<grid, 256, 0, stream>>>(
            A, lda, aS, Bmat, ldb, bS, A2, B2,
            Cf, Cb, ldc, transC, zDiv, cOuter, cInner, bias, act, K);
    };

    const long nelem = NTOK * DD;  // 2M
    init_kernel<<<(int)((nelem + 255) / 256), 256, 0, stream>>>(x_dec, x_enc, x, xbf, xebf);
    fill_ddft_t<<<(1024 * 512) / 256, 256, 0, stream>>>(DdftT);
    fill_cs<<<(1024 * 1024) / 256, 256, 0, stream>>>(Cs, NSs);

    for (int l = 0; l < LL; ++l) {
        // ---- FFT token mixing as DFT matmuls ----
        // Tt[q][token] = (x @ [C_D | S_D])^T : [4096,512] x (B^T [1024][512])
        gemm(xbf, 512, 0, DdftT, 512, 0, nullptr, nullptr,
             nullptr, Tt, 4096, /*transC=*/1, 1, 0, 0, nullptr, 0, 4096, 1024, 512, 1);
        // xf[b] = C_S @ T1[b] + (-S_S) @ T2[b]  (batched over b; B from Tt)
        gemm(Cs, 1024, 0, Tt, 4096, 1024, NSs, Tt + (long)512 * 4096,
             addb, nullptr, 512, 0, 1, (long)SS * DD, 0, nullptr, 0, 1024, 512, 1024, BB);
        ln_add_kernel<<<(int)NTOK, 256, 0, stream>>>(x, addb, ln1w + (long)l * DD, ln1b + (long)l * DD, x, xbf);

        // ---- per-head QKV projections ----
        pack_headw_t<<<8192, 256, 0, stream>>>(Wq + (long)l * HH * DD * DD, WqpT);
        pack_headw_t<<<8192, 256, 0, stream>>>(Wk + (long)l * HH * DD * DD, WkpT);
        pack_headw_t<<<8192, 256, 0, stream>>>(Wv + (long)l * HH * DD * DD, WvpT);

        gemm(xbf, 512, 0, WqpT, 512, 0, nullptr, nullptr,
             G, nullptr, 4096, 0, 1, 0, 0, bq + (long)l * HH * DD, 0, 4096, 4096, 512, 1);
        head_softmax<<<8192, 256, 0, stream>>>(G, Qbf, 1, 0);   // [b,h,s,e]
        gemm(xebf, 512, 0, WkpT, 512, 0, nullptr, nullptr,
             G, nullptr, 4096, 0, 1, 0, 0, bk + (long)l * HH * DD, 0, 4096, 4096, 512, 1);
        head_softmax<<<8192, 256, 0, stream>>>(G, KbfT, 1, 1);  // [b,h,e,s]
        gemm(xebf, 512, 0, WvpT, 512, 0, nullptr, nullptr,
             G, nullptr, 4096, 0, 1, 0, 0, bv + (long)l * HH * DD, 0, 4096, 4096, 512, 1);
        head_softmax<<<8192, 256, 0, stream>>>(G, VbfT, 0, 1);  // [b,h,e,s]

        // ---- gc = K^T V per (b,h): A=KbfT [d][s], B^T=VbfT [e][s]; out gc^T [e][d]
        gemm(KbfT, 1024, (long)SS * DD, VbfT, 1024, (long)SS * DD, nullptr, nullptr,
             nullptr, gcbfT, 512, /*transC=*/1, 1, (long)512 * 512, 0,
             nullptr, 0, 512, 512, 1024, BB * HH);
        // ---- att = Q @ gc per (b,h): A=Qbf [s][d], B^T=gc^T [e][d]; out [b,s,h*D+e]
        gemm(Qbf, 512, (long)SS * DD, gcbfT, 512, (long)512 * 512, nullptr, nullptr,
             nullptr, attbf, 4096, 0, HH, (long)SS * 4096, 512,
             nullptr, 0, 1024, 512, 512, BB * HH);

        // ---- output projection ----
        pack_wo_t<<<8192, 256, 0, stream>>>(Wo + (long)l * DD * HH * DD, WopT);
        gemm(attbf, 4096, 0, WopT, 4096, 0, nullptr, nullptr,
             addb, nullptr, 512, 0, 1, 0, 0, bo + (long)l * DD, 0, 4096, 512, 4096, 1);
        ln_add_kernel<<<(int)NTOK, 256, 0, stream>>>(x, addb, ln2w + (long)l * DD, ln2b + (long)l * DD, x, xbf);

        // ---- feed-forward (SELU fused in GEMM1 epilogue) ----
        pack_t512<<<1024, 256, 0, stream>>>(W1 + (long)l * DD * DD, W1pT);
        pack_t512<<<1024, 256, 0, stream>>>(W2 + (long)l * DD * DD, W2pT);
        gemm(xbf, 512, 0, W1pT, 512, 0, nullptr, nullptr,
             nullptr, h1bf, 512, 0, 1, 0, 0, b1 + (long)l * DD, 1, 4096, 512, 512, 1);
        gemm(h1bf, 512, 0, W2pT, 512, 0, nullptr, nullptr,
             addb, nullptr, 512, 0, 1, 0, 0, b2 + (long)l * DD, 0, 4096, 512, 512, 1);
        ln_add_kernel<<<(int)NTOK, 256, 0, stream>>>(x, addb, ln3w + (long)l * DD, ln3b + (long)l * DD, x, xbf);
    }

    copy_out_kernel<<<(int)((nelem + 255) / 256), 256, 0, stream>>>(x, (float*)d_out, nelem);
}
